// HSTGNN_65498251264551
// MI455X (gfx1250) — compile-verified, matrix operand order checked by reading, not run on hardware
//
#include <hip/hip_runtime.h>
#include <hip/hip_bf16.h>
#include <math.h>

typedef __attribute__((ext_vector_type(16))) _Float16 v16h;
typedef __attribute__((ext_vector_type(8)))  _Float16 v8h;
typedef __attribute__((ext_vector_type(8)))  float    v8f;

#define B_  16
#define N_  2048
#define T_  96
#define D_  32
#define M_  2144        // N_+T_
#define NT_ 128         // N_/16 tile boundary
#define MT_ 134         // M_/16 tiles per dim
#define SQRT3 1.7320508075688772f

// ---- workspace layout (bytes) ----
#define OFF_SA  ((size_t)0)                         // (B,N,96) f16  [s, √3 n1, −√3 n2]
#define OFF_SB  (OFF_SA + (size_t)B_*N_*96*2)       // (B,N,96) f16  [s, √3 n2,  √3 n1]
#define OFF_SQ  (OFF_SB + (size_t)B_*N_*96*2)       // (B,N,64) f16  [s, q_st]
#define OFF_SK  (OFF_SQ + (size_t)B_*N_*64*2)       // (B,N,64) f16  [s, k_ts]
#define OFF_TQ  (OFF_SK + (size_t)B_*N_*64*2)       // (B,T,64) f16  [t, q_ts]
#define OFF_TK  (OFF_TQ + (size_t)B_*T_*64*2)       // (B,T,64) f16  [t, k_st]
#define OFF_TT  (OFF_TK + (size_t)B_*T_*64*2)       // (B,T,32) f16  [t]
#define OFF_MAX (OFF_TT + (size_t)B_*T_*32*2)       // 4 x u32 (relu-max per block type)

__global__ void init_max_kernel(unsigned int* m) {
    if (threadIdx.x < 4) m[threadIdx.x] = 0u;       // bits of +0.0f
}

// -------------------- feature precompute --------------------
// one thread per (batch, node); weights broadcast from LDS
__global__ __launch_bounds__(256) void feat_kernel(
    const float* __restrict__ s, const float* __restrict__ t,
    const float* __restrict__ Wss1, const float* __restrict__ Wss2,
    const float* __restrict__ Wqst, const float* __restrict__ bqst,
    const float* __restrict__ Wkst, const float* __restrict__ bkst,
    const float* __restrict__ Wqts, const float* __restrict__ bqts,
    const float* __restrict__ Wkts, const float* __restrict__ bkts,
    _Float16* __restrict__ SA, _Float16* __restrict__ SB,
    _Float16* __restrict__ SQ, _Float16* __restrict__ SK,
    _Float16* __restrict__ TQ, _Float16* __restrict__ TK,
    _Float16* __restrict__ TT)
{
    __shared__ float lw[6272];   // 6x 32x32 mats + 4x 32 biases
    const int tid = threadIdx.x;
    for (int i = tid; i < 6272; i += 256) {
        float v;
        if      (i < 1024) v = Wss1[i];
        else if (i < 2048) v = Wss2[i - 1024];
        else if (i < 3072) v = Wqst[i - 2048];
        else if (i < 4096) v = Wkst[i - 3072];
        else if (i < 5120) v = Wqts[i - 4096];
        else if (i < 6144) v = Wkts[i - 5120];
        else if (i < 6176) v = bqst[i - 6144];
        else if (i < 6208) v = bkst[i - 6176];
        else if (i < 6240) v = bqts[i - 6208];
        else               v = bkts[i - 6240];
        lw[i] = v;
    }
    __syncthreads();

    const int g = blockIdx.x * 256 + tid;
    if (g >= B_ * M_) return;
    const int b = g / M_;
    const int node = g % M_;

    float x[D_];
    if (node < N_) {
        const float* xr = s + ((size_t)b * N_ + node) * D_;
        #pragma unroll
        for (int k = 0; k < D_; ++k) x[k] = xr[k];

        _Float16* sa = SA + ((size_t)b * N_ + node) * 96;
        _Float16* sb = SB + ((size_t)b * N_ + node) * 96;
        _Float16* sq = SQ + ((size_t)b * N_ + node) * 64;
        _Float16* sk = SK + ((size_t)b * N_ + node) * 64;
        #pragma unroll
        for (int k = 0; k < D_; ++k) {
            _Float16 h = (_Float16)x[k];
            sa[k] = h; sb[k] = h; sq[k] = h; sk[k] = h;
        }
        for (int j = 0; j < D_; ++j) {
            float d1 = 0.f, d2 = 0.f, dq = 0.f, dk = 0.f;
            #pragma unroll
            for (int k = 0; k < D_; ++k) {
                const float xv = x[k];
                d1 += xv * lw[      j * 32 + k];   // W_ss1
                d2 += xv * lw[1024 + j * 32 + k];  // W_ss2
                dq += xv * lw[2048 + j * 32 + k];  // Wq_st
                dk += xv * lw[5120 + j * 32 + k];  // Wk_ts
            }
            const float n1 = tanhf(3.f * d1);
            const float n2 = tanhf(3.f * d2);
            sa[32 + j] = (_Float16)( SQRT3 * n1);
            sa[64 + j] = (_Float16)(-SQRT3 * n2);
            sb[32 + j] = (_Float16)( SQRT3 * n2);
            sb[64 + j] = (_Float16)( SQRT3 * n1);
            sq[32 + j] = (_Float16)(dq + lw[6144 + j]);  // + bq_st
            sk[32 + j] = (_Float16)(dk + lw[6240 + j]);  // + bk_ts
        }
    } else {
        const int nt = node - N_;
        const float* xr = t + ((size_t)b * T_ + nt) * D_;
        #pragma unroll
        for (int k = 0; k < D_; ++k) x[k] = xr[k];

        _Float16* tq = TQ + ((size_t)b * T_ + nt) * 64;
        _Float16* tk = TK + ((size_t)b * T_ + nt) * 64;
        _Float16* tt = TT + ((size_t)b * T_ + nt) * 32;
        #pragma unroll
        for (int k = 0; k < D_; ++k) {
            _Float16 h = (_Float16)x[k];
            tq[k] = h; tk[k] = h; tt[k] = h;
        }
        for (int j = 0; j < D_; ++j) {
            float dq = 0.f, dk = 0.f;
            #pragma unroll
            for (int k = 0; k < D_; ++k) {
                const float xv = x[k];
                dq += xv * lw[4096 + j * 32 + k];  // Wq_ts
                dk += xv * lw[3072 + j * 32 + k];  // Wk_st
            }
            tq[32 + j] = (_Float16)(dq + lw[6208 + j]);  // + bq_ts
            tk[32 + j] = (_Float16)(dk + lw[6176 + j]);  // + bk_st
        }
    }
}

// -------------------- WMMA adjacency GEMM + block max --------------------
// grid: (17 col-strips, 134 tile-rows, 16 batches); 8 waves/block, 1 tile/wave
__global__ __launch_bounds__(256) void adj_wmma_kernel(
    const _Float16* __restrict__ SA, const _Float16* __restrict__ SB,
    const _Float16* __restrict__ SQ, const _Float16* __restrict__ SK,
    const _Float16* __restrict__ TQ, const _Float16* __restrict__ TK,
    const _Float16* __restrict__ TT,
    float* __restrict__ out, unsigned int* __restrict__ maxbuf)
{
    const int lane = threadIdx.x & 31;
    const int w    = threadIdx.x >> 5;
    const int tRow = blockIdx.y;
    const int tCol = blockIdx.x * 8 + w;
    const int b    = blockIdx.z;
    const bool rowS = (tRow < NT_);
    const bool colS = (blockIdx.x < 16);          // uniform per block (128 % 8 == 0)
    const int  btype = (rowS ? 0 : 2) + (colS ? 0 : 1);

    float lmax = 0.f;
    if (tCol < MT_) {
        const _Float16 *Ab, *Bb;
        int K;
        if (rowS) {
            if (colS) { K = 96; Ab = SA + ((size_t)b*N_ + tRow*16)*96;        Bb = SB + ((size_t)b*N_ + tCol*16)*96; }
            else      { K = 64; Ab = SQ + ((size_t)b*N_ + tRow*16)*64;        Bb = TK + ((size_t)b*T_ + (tCol-NT_)*16)*64; }
        } else {
            if (colS) { K = 64; Ab = TQ + ((size_t)b*T_ + (tRow-NT_)*16)*64;  Bb = SK + ((size_t)b*N_ + tCol*16)*64; }
            else      { K = 32; Ab = TT + ((size_t)b*T_ + (tRow-NT_)*16)*32;  Bb = TT + ((size_t)b*T_ + (tCol-NT_)*16)*32; }
        }
        const int node = lane & 15;               // row (A) / col (B) within tile
        const int koff = (lane >> 4) << 3;        // lane<16: K 0-7 & 16-23; lane>=16: K 8-15 & 24-31

        v8f c = {};
        for (int kc = 0; kc < K; kc += 32) {
            const _Float16* pa = Ab + node * K + kc + koff;
            const _Float16* pb = Bb + node * K + kc + koff;
            v8h a0 = *(const v8h*)pa;  v8h a1 = *(const v8h*)(pa + 16);
            v8h b0 = *(const v8h*)pb;  v8h b1 = *(const v8h*)(pb + 16);
            v16h a, bm;
            #pragma unroll
            for (int i = 0; i < 8; ++i) { a[i] = a0[i]; a[i+8] = a1[i]; bm[i] = b0[i]; bm[i+8] = b1[i]; }
            c = __builtin_amdgcn_wmma_f32_16x16x32_f16(false, a, false, bm, (short)0, c,
                                                       false, false);
        }

        const int col     = tCol * 16 + node;
        const int rowbase = tRow * 16 + ((lane >> 4) << 3);
        const size_t obase = (size_t)b * M_ * M_ + (size_t)rowbase * M_ + col;
        #pragma unroll
        for (int v = 0; v < 8; ++v) {
            const float val = c[v];
            out[obase + (size_t)v * M_] = val;
            lmax = fmaxf(lmax, val);              // lmax starts at 0 => relu-max
        }
    }

    // wave32 reduce, then block reduce, one atomic per block
    #pragma unroll
    for (int off = 16; off > 0; off >>= 1)
        lmax = fmaxf(lmax, __shfl_xor(lmax, off, 32));
    __shared__ float wred[8];
    if (lane == 0) wred[w] = lmax;
    __syncthreads();
    if (threadIdx.x == 0) {
        float m = wred[0];
        #pragma unroll
        for (int i = 1; i < 8; ++i) m = fmaxf(m, wred[i]);
        atomicMax(maxbuf + btype, __float_as_uint(m));  // all values >= 0: uint order == float order
    }
}

// -------------------- normalize: tanh(relu(x)/(max+eps)), triu on tt --------------------
// grid: (M_*M_/4/256 = 4489, B_); float4 per thread
__global__ __launch_bounds__(256) void norm_kernel(float* __restrict__ out,
                                                   const float* __restrict__ maxbuf)
{
    const int b = blockIdx.y;
    const size_t e = ((size_t)blockIdx.x * 256 + threadIdx.x) * 4;
    const int r  = (int)(e / M_);
    const int c0 = (int)(e % M_);
    const int btype = ((r < N_) ? 0 : 2) + ((c0 < N_) ? 0 : 1);  // 4 cols share type (N_%4==0)
    const float div = maxbuf[btype];
    const float inv = 1.f / (div + 1e-30f);

    float* p = out + (size_t)b * M_ * M_ + e;
    float4 v = *(float4*)p;
    float vals[4] = {v.x, v.y, v.z, v.w};
    #pragma unroll
    for (int i = 0; i < 4; ++i) {
        float rr = fmaxf(vals[i], 0.f);
        float tv = (div > 0.f) ? tanhf(rr * inv) : tanhf(rr);
        if (btype == 3 && r > (c0 + i)) tv = 0.f;  // triu: keep local row <= local col
        vals[i] = tv;
    }
    *(float4*)p = make_float4(vals[0], vals[1], vals[2], vals[3]);
}

extern "C" void kernel_launch(void* const* d_in, const int* in_sizes, int n_in,
                              void* d_out, int out_size, void* d_ws, size_t ws_size,
                              hipStream_t stream) {
    const float* s    = (const float*)d_in[0];
    const float* t    = (const float*)d_in[1];
    const float* Wss1 = (const float*)d_in[2];
    const float* Wss2 = (const float*)d_in[3];
    const float* Wqst = (const float*)d_in[4];
    const float* bqst = (const float*)d_in[5];
    const float* Wkst = (const float*)d_in[6];
    const float* bkst = (const float*)d_in[7];
    const float* Wqts = (const float*)d_in[8];
    const float* bqts = (const float*)d_in[9];
    const float* Wkts = (const float*)d_in[10];
    const float* bkts = (const float*)d_in[11];
    float* out = (float*)d_out;
    char* ws = (char*)d_ws;

    _Float16* SA = (_Float16*)(ws + OFF_SA);
    _Float16* SB = (_Float16*)(ws + OFF_SB);
    _Float16* SQ = (_Float16*)(ws + OFF_SQ);
    _Float16* SK = (_Float16*)(ws + OFF_SK);
    _Float16* TQ = (_Float16*)(ws + OFF_TQ);
    _Float16* TK = (_Float16*)(ws + OFF_TK);
    _Float16* TT = (_Float16*)(ws + OFF_TT);
    unsigned int* maxbuf = (unsigned int*)(ws + OFF_MAX);

    init_max_kernel<<<1, 32, 0, stream>>>(maxbuf);

    feat_kernel<<<(B_ * M_ + 255) / 256, 256, 0, stream>>>(
        s, t, Wss1, Wss2, Wqst, bqst, Wkst, bkst, Wqts, bqts, Wkts, bkts,
        SA, SB, SQ, SK, TQ, TK, TT);

    adj_wmma_kernel<<<dim3(17, MT_, B_), 256, 0, stream>>>(
        SA, SB, SQ, SK, TQ, TK, TT, out, maxbuf);

    norm_kernel<<<dim3((M_ * M_ / 4) / 256, B_), 256, 0, stream>>>(
        out, (const float*)maxbuf);
}
